// TriggerModel_12695923327130
// MI455X (gfx1250) — matrix operand order repaired and checked by reading, not compiled, not run on hardware
//
#include <hip/hip_runtime.h>

// ---------------- model constants ----------------
constexpr int kL = 12, kNH = 12, kDH = 64, kH = 768, kFF = 3072;
constexpr int kB = 16, kS = 256, kMID = 256;
constexpr int kEV = 33, kC = 2 * kEV + 1 + 2;      // 69 tags
constexpr int kSTART = kC - 2, kSTOP = kC - 1;     // 67, 68

typedef __attribute__((ext_vector_type(16))) __bf16 bf16x16;
typedef __attribute__((ext_vector_type(8)))  __bf16 bf16x8;
typedef __attribute__((ext_vector_type(8)))  float  f32x8;

// ---------------- small device helpers ----------------
__device__ __forceinline__ float gelu_tanh(float x) {
  const float c0 = 0.7978845608028654f, c1 = 0.044715f;
  float x3 = x * x * x;
  return 0.5f * x * (1.0f + tanhf(c0 * (x + c1 * x3)));
}

__device__ __forceinline__ bf16x8 cvt8(float4 a, float4 b) {
  bf16x8 r;
  r[0] = (__bf16)a.x; r[1] = (__bf16)a.y; r[2] = (__bf16)a.z; r[3] = (__bf16)a.w;
  r[4] = (__bf16)b.x; r[5] = (__bf16)b.y; r[6] = (__bf16)b.z; r[7] = (__bf16)b.w;
  return r;
}

// block reductions for blockDim == 256 (8 wave32 waves)
__device__ __forceinline__ float block_sum256(float v, float* sbuf) {
#pragma unroll
  for (int o = 16; o > 0; o >>= 1) v += __shfl_down(v, o, 32);
  int lane = threadIdx.x & 31, w = threadIdx.x >> 5;
  if (lane == 0) sbuf[w] = v;
  __syncthreads();
  float r = (threadIdx.x < 8) ? sbuf[threadIdx.x] : 0.0f;
  if (w == 0) {
#pragma unroll
    for (int o = 4; o > 0; o >>= 1) r += __shfl_down(r, o, 32);
    if (lane == 0) sbuf[0] = r;
  }
  __syncthreads();
  r = sbuf[0];
  __syncthreads();
  return r;
}

__device__ __forceinline__ float block_max256(float v, float* sbuf) {
#pragma unroll
  for (int o = 16; o > 0; o >>= 1) v = fmaxf(v, __shfl_down(v, o, 32));
  int lane = threadIdx.x & 31, w = threadIdx.x >> 5;
  if (lane == 0) sbuf[w] = v;
  __syncthreads();
  float r = (threadIdx.x < 8) ? sbuf[threadIdx.x] : -3.0e38f;
  if (w == 0) {
#pragma unroll
    for (int o = 4; o > 0; o >>= 1) r = fmaxf(r, __shfl_down(r, o, 32));
    if (lane == 0) sbuf[0] = r;
  }
  __syncthreads();
  r = sbuf[0];
  __syncthreads();
  return r;
}

// ---------------- generic batched bf16 WMMA GEMM ----------------
// C[z] = act(alpha * A[z] @ op(B[z]) + bias + kbias), z = zb*nh + zh
// Requirements (all call sites satisfy): M % 64 == 0, K % 32 == 0, lda % 4 == 0.
// Block tile 64x64, K step 32, 8 waves = 4(M) x 2(N-pairs); each wave:
// one 16x32 A-frag x two 32x16 B-frags -> two v_wmma_f32_16x16x32_bf16.
template <int TRANSB>
__global__ __launch_bounds__(256) void gemm_bf16_kernel(
    const float* __restrict__ A, const float* __restrict__ Bw,
    float* __restrict__ Cm,
    const float* __restrict__ bias,   // per output col, may be null
    const float* __restrict__ kbias,  // per (zb, col) additive, may be null
    int M, int N, int K, int lda, int ldb, int ldc,
    long long sAb, long long sAh, long long sBb, long long sBh,
    long long sCb, long long sCh, int nh, int act, float alpha) {
  const int z = blockIdx.z;
  const int zb = z / nh, zh = z - zb * nh;
  A  += (long long)zb * sAb + (long long)zh * sAh;
  Bw += (long long)zb * sBb + (long long)zh * sBh;
  Cm += (long long)zb * sCb + (long long)zh * sCh;

  // K-contiguous tiles; stride 40 bf16 = 80B keeps 16B alignment per sub-row.
  __shared__ __align__(16) __bf16 As [64][40];  // [m][k]
  __shared__ __align__(16) __bf16 BsT[64][40];  // [n][k]

  const int tid  = threadIdx.x;
  const int wave = tid >> 5;
  const int lane = tid & 31;
  const int wm   = wave & 3;   // M sub-tile
  const int wn   = wave >> 2;  // N sub-tile pair {2wn, 2wn+1}
  const int hi   = lane >> 4;  // lane half
  const int lm   = lane & 15;

  const int m0 = blockIdx.y * 64;
  const int n0 = blockIdx.x * 64;

  // staging thread mapping
  const int ra = tid >> 2, ca = (tid & 3) * 8;          // A: 64 rows x 32 cols
  const int rb = tid >> 2, cb = (tid & 3) * 8;          // B trans: n-row, k-col
  const int rk = tid >> 3, cn = (tid & 7) * 8;          // B normal: k-row, n-col
  const bool nfull = (n0 + 64 <= N);
  const bool bfast = TRANSB ? nfull : (nfull && ((ldb & 3) == 0));

  f32x8 acc0 = {};
  f32x8 acc1 = {};

  float4 ar0, ar1, br0, br1;

  auto load_tile = [&](int k0) {
    const float* ap = A + (long long)(m0 + ra) * lda + (k0 + ca);
    ar0 = *(const float4*)ap;
    ar1 = *(const float4*)(ap + 4);
    if (TRANSB) {
      int gn = n0 + rb;
      int gc = (gn < N) ? gn : (N - 1);
      const float* bp = Bw + (long long)gc * ldb + (k0 + cb);
      br0 = *(const float4*)bp;
      br1 = *(const float4*)(bp + 4);
      if (!bfast && gn >= N) {
        br0 = make_float4(0.f, 0.f, 0.f, 0.f);
        br1 = br0;
      }
    } else if (bfast) {
      const float* bp = Bw + (long long)(k0 + rk) * ldb + (n0 + cn);
      br0 = *(const float4*)bp;
      br1 = *(const float4*)(bp + 4);
    } else {
      const float* bp = Bw + (long long)(k0 + rk) * ldb;
      float t[8];
#pragma unroll
      for (int j = 0; j < 8; ++j) {
        int gn = n0 + cn + j;
        int gc = (gn < N) ? gn : (N - 1);
        float v = bp[gc];
        t[j] = (gn < N) ? v : 0.0f;
      }
      br0 = make_float4(t[0], t[1], t[2], t[3]);
      br1 = make_float4(t[4], t[5], t[6], t[7]);
    }
  };

  load_tile(0);  // prologue

  for (int k0 = 0; k0 < K; k0 += 32) {
    // commit staged registers to LDS (wide stores)
    *(bf16x8*)&As[ra][ca] = cvt8(ar0, ar1);
    if (TRANSB) {
      *(bf16x8*)&BsT[rb][cb] = cvt8(br0, br1);
    } else {
      bf16x8 bv = cvt8(br0, br1);
#pragma unroll
      for (int j = 0; j < 8; ++j) BsT[cn + j][rk] = bv[j];  // transpose-on-store
    }
    __syncthreads();

    if (k0 + 32 < K) load_tile(k0 + 32);  // overlap next tile with math
    if (k0 + 64 < K)                       // hint -> global_prefetch_b8
      __builtin_prefetch(A + (long long)(m0 + ra) * lda + k0 + 64, 0, 1);

    // A fragment: rows wm*16+lm; K run {hi*16..+15}
    const __bf16* ap = &As[wm * 16 + lm][hi * 16];
    bf16x8 alo = *(const bf16x8*)ap;
    bf16x8 ahi = *(const bf16x8*)(ap + 8);
    bf16x16 af = __builtin_shufflevector(alo, ahi, 0, 1, 2, 3, 4, 5, 6, 7, 8,
                                         9, 10, 11, 12, 13, 14, 15);
    // B fragments: col n, K run {hi*16..+15}
    const __bf16* bp0 = &BsT[(wn * 2 + 0) * 16 + lm][hi * 16];
    bf16x8 b0l = *(const bf16x8*)bp0;
    bf16x8 b0h = *(const bf16x8*)(bp0 + 8);
    bf16x16 bf0 = __builtin_shufflevector(b0l, b0h, 0, 1, 2, 3, 4, 5, 6, 7, 8,
                                          9, 10, 11, 12, 13, 14, 15);
    const __bf16* bp1 = &BsT[(wn * 2 + 1) * 16 + lm][hi * 16];
    bf16x8 b1l = *(const bf16x8*)bp1;
    bf16x8 b1h = *(const bf16x8*)(bp1 + 8);
    bf16x16 bf1 = __builtin_shufflevector(b1l, b1h, 0, 1, 2, 3, 4, 5, 6, 7, 8,
                                          9, 10, 11, 12, 13, 14, 15);

    acc0 = __builtin_amdgcn_wmma_f32_16x16x32_bf16(false, af, false, bf0,
                                                   (short)0, acc0, false, false);
    acc1 = __builtin_amdgcn_wmma_f32_16x16x32_bf16(false, af, false, bf1,
                                                   (short)0, acc1, false, false);
    __syncthreads();
  }

  // epilogue: C/D layout — VGPR v holds row v + 8*hi, col = lane&15
#pragma unroll
  for (int t = 0; t < 2; ++t) {
    f32x8 a = t ? acc1 : acc0;
    int n = n0 + (wn * 2 + t) * 16 + lm;
    if (n < N) {
      float bv = bias ? bias[n] : 0.0f;
      if (kbias) bv += kbias[(long long)zb * N + n];
#pragma unroll
      for (int v = 0; v < 8; ++v) {
        int m = m0 + wm * 16 + (hi << 3) + v;
        float r = alpha * a[v] + bv;
        if (act == 1) r = gelu_tanh(r);
        else if (act == 2) r = fmaxf(r, 0.0f);
        Cm[(long long)m * ldc + n] = r;
      }
    }
  }
}

// ---------------- embeddings + LayerNorm ----------------
__global__ __launch_bounds__(256) void embed_ln_kernel(
    const int* __restrict__ ids, const int* __restrict__ tt,
    const float* __restrict__ we, const float* __restrict__ pe,
    const float* __restrict__ te, const float* __restrict__ g,
    const float* __restrict__ b, float* __restrict__ x) {
  __shared__ float sbuf[8];
  const int token = blockIdx.x;            // b*S + s
  const int s = token % kS;
  const int id = ids[token];
  const int ty = tt[token];
  float e[3], sum = 0.0f;
#pragma unroll
  for (int i = 0; i < 3; ++i) {
    int d = threadIdx.x + i * 256;
    e[i] = we[(long long)id * kH + d] + pe[(long long)s * kH + d] +
           te[(long long)ty * kH + d];
    sum += e[i];
  }
  float mean = block_sum256(sum, sbuf) * (1.0f / kH);
  float var = 0.0f;
#pragma unroll
  for (int i = 0; i < 3; ++i) { float d0 = e[i] - mean; var += d0 * d0; }
  var = block_sum256(var, sbuf) * (1.0f / kH);
  float inv = rsqrtf(var + 1e-12f);
#pragma unroll
  for (int i = 0; i < 3; ++i) {
    int d = threadIdx.x + i * 256;
    x[(long long)token * kH + d] = (e[i] - mean) * inv * g[d] + b[d];
  }
}

// x = LN(x + y) * g + b   (row-wise, in place on x)
__global__ __launch_bounds__(256) void residual_ln_kernel(
    float* __restrict__ x, const float* __restrict__ y,
    const float* __restrict__ g, const float* __restrict__ b) {
  __shared__ float sbuf[8];
  const long long row = blockIdx.x;
  float e[3], sum = 0.0f;
#pragma unroll
  for (int i = 0; i < 3; ++i) {
    int d = threadIdx.x + i * 256;
    e[i] = x[row * kH + d] + y[row * kH + d];
    sum += e[i];
  }
  float mean = block_sum256(sum, sbuf) * (1.0f / kH);
  float var = 0.0f;
#pragma unroll
  for (int i = 0; i < 3; ++i) { float d0 = e[i] - mean; var += d0 * d0; }
  var = block_sum256(var, sbuf) * (1.0f / kH);
  float inv = rsqrtf(var + 1e-12f);
#pragma unroll
  for (int i = 0; i < 3; ++i) {
    int d = threadIdx.x + i * 256;
    x[row * kH + d] = (e[i] - mean) * inv * g[d] + b[d];
  }
}

// additive attention bias per key position: (1-mask)*-1e4
__global__ void maskbias_kernel(const int* __restrict__ mask,
                                float* __restrict__ mb, int n) {
  int i = blockIdx.x * blockDim.x + threadIdx.x;
  if (i < n) mb[i] = (1.0f - (float)mask[i]) * -10000.0f;
}

// row softmax over 256 columns (one thread per column)
__global__ __launch_bounds__(256) void softmax_kernel(float* __restrict__ sc) {
  __shared__ float sbuf[8];
  float* r = sc + (long long)blockIdx.x * kS;
  float v = r[threadIdx.x];
  float m = block_max256(v, sbuf);
  float e = expf(v - m);
  float s = block_sum256(e, sbuf);
  r[threadIdx.x] = e / s;
}

__global__ void zero_loss_kernel(float* p) { *p = 0.0f; }

// ---------------- CRF forward (NLL) ----------------
__global__ __launch_bounds__(128) void crf_forward_kernel(
    const float* __restrict__ feats, const int* __restrict__ mask,
    const int* __restrict__ labels, const float* __restrict__ trans,
    float* __restrict__ loss) {
  __shared__ float sT[kC * kC];
  __shared__ float alpha[2][kC];
  __shared__ float red[kC];
  const int b = blockIdx.x, tid = threadIdx.x;
  for (int i = tid; i < kC * kC; i += 128) sT[i] = trans[i];
  __syncthreads();
  const float* fb = feats + (long long)b * kS * kC;
  const int* mrow = mask + b * kS;
  const int* lb = labels + b * kS;

  int p = 0;
  if (tid < kC) alpha[0][tid] = sT[kSTART * kC + tid] + fb[tid];
  __syncthreads();
  for (int t = 1; t < kS; ++t) {
    float nj = 0.0f;
    if (tid < kC) {
      float mx = -3.0e38f;
      for (int i = 0; i < kC; ++i)
        mx = fmaxf(mx, alpha[p][i] + sT[i * kC + tid]);
      float sm = 0.0f;
      for (int i = 0; i < kC; ++i)
        sm += expf(alpha[p][i] + sT[i * kC + tid] - mx);
      nj = mx + logf(sm) + fb[(long long)t * kC + tid];
    }
    int mt = mrow[t];
    __syncthreads();
    if (tid < kC) alpha[1 - p][tid] = (mt > 0) ? nj : alpha[p][tid];
    __syncthreads();
    p = 1 - p;
  }
  if (tid < kC) red[tid] = alpha[p][tid] + sT[tid * kC + kSTOP];
  __syncthreads();
  if (tid == 0) {
    float mx = -3.0e38f;
    for (int i = 0; i < kC; ++i) mx = fmaxf(mx, red[i]);
    float sm = 0.0f;
    for (int i = 0; i < kC; ++i) sm += expf(red[i] - mx);
    float logZ = mx + logf(sm);
    float emit = 0.0f, pair = 0.0f;
    int len = 0;
    for (int t = 0; t < kS; ++t)
      if (mrow[t] > 0) { emit += fb[(long long)t * kC + lb[t]]; ++len; }
    for (int t = 1; t < kS; ++t)
      if (mrow[t] > 0) pair += sT[lb[t - 1] * kC + lb[t]];
    int last = lb[len - 1];
    float gold = emit + pair + sT[kSTART * kC + lb[0]] + sT[last * kC + kSTOP];
    atomicAdd(loss, logZ - gold);
  }
}

// ---------------- Viterbi decode ----------------
__global__ __launch_bounds__(128) void viterbi_kernel(
    const float* __restrict__ feats, const int* __restrict__ mask,
    const float* __restrict__ trans, int* __restrict__ bp,
    float* __restrict__ path) {
  __shared__ float sT[kC * kC];
  __shared__ float vb[2][kC];
  const int b = blockIdx.x, tid = threadIdx.x;
  for (int i = tid; i < kC * kC; i += 128) sT[i] = trans[i];
  __syncthreads();
  const float* fb = feats + (long long)b * kS * kC;
  const int* mrow = mask + b * kS;
  int* bpb = bp + (long long)b * kS * kC;

  int p = 0;
  if (tid < kC) vb[0][tid] = sT[kSTART * kC + tid] + fb[tid];
  __syncthreads();
  for (int t = 1; t < kS; ++t) {
    float nv = 0.0f;
    int bi = 0;
    if (tid < kC) {
      float best = -3.0e38f;
      for (int i = 0; i < kC; ++i) {
        float sc = vb[p][i] + sT[i * kC + tid];
        if (sc > best) { best = sc; bi = i; }   // first-max tie break
      }
      nv = best + fb[(long long)t * kC + tid];
    }
    int mt = mrow[t];
    __syncthreads();
    if (tid < kC) {
      vb[1 - p][tid] = (mt > 0) ? nv : vb[p][tid];
      bpb[(long long)t * kC + tid] = (mt > 0) ? bi : tid;
    }
    __syncthreads();
    p = 1 - p;
  }
  if (tid == 0) {
    float best = -3.0e38f;
    int tag = 0;
    for (int j = 0; j < kC; ++j) {
      float sc = vb[p][j] + sT[j * kC + kSTOP];
      if (sc > best) { best = sc; tag = j; }
    }
    path[(long long)b * kS + (kS - 1)] = (float)tag;
    for (int t = kS - 1; t >= 1; --t) {
      tag = bpb[(long long)t * kC + tag];
      path[(long long)b * kS + t - 1] = (float)tag;
    }
  }
}

// ---------------- host launcher ----------------
static void launch_gemm(hipStream_t stream, const float* A, const float* Bw,
                        float* Cm, const float* bias, const float* kbias,
                        int M, int N, int K, int lda, int ldb, int ldc,
                        long long sAb, long long sAh, long long sBb,
                        long long sBh, long long sCb, long long sCh, int nh,
                        int Z, int transB, int act, float alpha) {
  dim3 grid((N + 63) / 64, (M + 63) / 64, Z);
  if (transB)
    gemm_bf16_kernel<1><<<grid, 256, 0, stream>>>(
        A, Bw, Cm, bias, kbias, M, N, K, lda, ldb, ldc, sAb, sAh, sBb, sBh,
        sCb, sCh, nh, act, alpha);
  else
    gemm_bf16_kernel<0><<<grid, 256, 0, stream>>>(
        A, Bw, Cm, bias, kbias, M, N, K, lda, ldb, ldc, sAb, sAh, sBb, sBh,
        sCb, sCh, nh, act, alpha);
}

extern "C" void kernel_launch(void* const* d_in, const int* in_sizes, int n_in,
                              void* d_out, int out_size, void* d_ws,
                              size_t ws_size, hipStream_t stream) {
  (void)in_sizes; (void)n_in; (void)out_size; (void)ws_size;
  const int*   input_ids = (const int*)d_in[0];
  const int*   attn_mask = (const int*)d_in[1];
  const int*   labels    = (const int*)d_in[2];
  const int*   type_ids  = (const int*)d_in[3];
  const float* word_emb  = (const float*)d_in[4];
  const float* pos_emb   = (const float*)d_in[5];
  const float* type_emb  = (const float*)d_in[6];
  const float* emb_g     = (const float*)d_in[7];
  const float* emb_b     = (const float*)d_in[8];
  const float* wq = (const float*)d_in[9];
  const float* wk = (const float*)d_in[10];
  const float* wv = (const float*)d_in[11];
  const float* wo = (const float*)d_in[12];
  const float* bq = (const float*)d_in[13];
  const float* bk = (const float*)d_in[14];
  const float* bv = (const float*)d_in[15];
  const float* bo = (const float*)d_in[16];
  const float* ln1g = (const float*)d_in[17];
  const float* ln1b = (const float*)d_in[18];
  const float* wff1 = (const float*)d_in[19];
  const float* bff1 = (const float*)d_in[20];
  const float* wff2 = (const float*)d_in[21];
  const float* bff2 = (const float*)d_in[22];
  const float* ln2g = (const float*)d_in[23];
  const float* ln2b = (const float*)d_in[24];
  const float* fcw1 = (const float*)d_in[25];
  const float* fcb1 = (const float*)d_in[26];
  const float* fcw2 = (const float*)d_in[27];
  const float* fcb2 = (const float*)d_in[28];
  const float* trans = (const float*)d_in[29];

  float* out = (float*)d_out;
  float* ws = (float*)d_ws;
  size_t off = 0;
  auto alloc = [&](size_t n) { float* p = ws + off; off += n; return p; };
  const int M = kB * kS;                       // 4096 rows
  float* x    = alloc((size_t)M * kH);         // residual stream
  float* y    = alloc((size_t)M * kH);         // gemm temp / ctx-proj / fc1
  float* q    = alloc((size_t)M * kH);         // Q, later attention context
  float* kbuf = alloc((size_t)M * kH);         // K
  float* vbuf = alloc((size_t)M * kH);         // V
  float* big  = alloc((size_t)M * kFF);        // scores (B*NH*S*S) == FFN hidden
  float* feats = alloc((size_t)M * kC);
  float* mb   = alloc((size_t)M);
  int*   bp   = (int*)alloc((size_t)M * kC);

  embed_ln_kernel<<<M, 256, 0, stream>>>(input_ids, type_ids, word_emb, pos_emb,
                                         type_emb, emb_g, emb_b, x);
  maskbias_kernel<<<(M + 255) / 256, 256, 0, stream>>>(attn_mask, mb, M);

  const long long HH = (long long)kH * kH;
  for (int l = 0; l < kL; ++l) {
    // Q/K/V projections
    launch_gemm(stream, x, wq + l * HH, q,    bq + l * kH, nullptr, M, kH, kH,
                kH, kH, kH, 0, 0, 0, 0, 0, 0, 1, 1, 0, 0, 1.0f);
    launch_gemm(stream, x, wk + l * HH, kbuf, bk + l * kH, nullptr, M, kH, kH,
                kH, kH, kH, 0, 0, 0, 0, 0, 0, 1, 1, 0, 0, 1.0f);
    launch_gemm(stream, x, wv + l * HH, vbuf, bv + l * kH, nullptr, M, kH, kH,
                kH, kH, kH, 0, 0, 0, 0, 0, 0, 1, 1, 0, 0, 1.0f);
    // scores[b,h] = Q_bh (S x DH) @ K_bh^T (DH x S) * 1/8 + maskbias
    launch_gemm(stream, q, kbuf, big, nullptr, mb, kS, kS, kDH,
                kH, kH, kS,
                (long long)kS * kH, kDH,                 // A per (b,h)
                (long long)kS * kH, kDH,                 // B per (b,h)
                (long long)kNH * kS * kS, (long long)kS * kS,
                kNH, kB * kNH, /*transB=*/1, 0, 0.125f);
    softmax_kernel<<<kB * kNH * kS, 256, 0, stream>>>(big);
    // ctx[b,h] = P (S x S) @ V_bh (S x DH)  -> overwrite q as [B,S,H]
    launch_gemm(stream, big, vbuf, q, nullptr, nullptr, kS, kDH, kS,
                kS, kH, kH,
                (long long)kNH * kS * kS, (long long)kS * kS,
                (long long)kS * kH, kDH,
                (long long)kS * kH, kDH,
                kNH, kB * kNH, 0, 0, 1.0f);
    // output projection + residual LN
    launch_gemm(stream, q, wo + l * HH, y, bo + l * kH, nullptr, M, kH, kH,
                kH, kH, kH, 0, 0, 0, 0, 0, 0, 1, 1, 0, 0, 1.0f);
    residual_ln_kernel<<<M, 256, 0, stream>>>(x, y, ln1g + l * kH, ln1b + l * kH);
    // FFN
    launch_gemm(stream, x, wff1 + (long long)l * kH * kFF, big, bff1 + l * kFF,
                nullptr, M, kFF, kH, kH, kFF, kFF, 0, 0, 0, 0, 0, 0, 1, 1, 0,
                /*gelu*/ 1, 1.0f);
    launch_gemm(stream, big, wff2 + (long long)l * kFF * kH, y, bff2 + l * kH,
                nullptr, M, kH, kFF, kFF, kH, kH, 0, 0, 0, 0, 0, 0, 1, 1, 0,
                0, 1.0f);
    residual_ln_kernel<<<M, 256, 0, stream>>>(x, y, ln2g + l * kH, ln2b + l * kH);
  }

  // head: relu(x @ fc_w1 + b1) @ fc_w2 + b2
  launch_gemm(stream, x, fcw1, y, fcb1, nullptr, M, kMID, kH, kH, kMID, kMID,
              0, 0, 0, 0, 0, 0, 1, 1, 0, /*relu*/ 2, 1.0f);
  launch_gemm(stream, y, fcw2, feats, fcb2, nullptr, M, kC, kMID, kMID, kC, kC,
              0, 0, 0, 0, 0, 0, 1, 1, 0, 0, 1.0f);

  zero_loss_kernel<<<1, 1, 0, stream>>>(out + (size_t)kB * kS);
  crf_forward_kernel<<<kB, 128, 0, stream>>>(feats, attn_mask, labels, trans,
                                             out + (size_t)kB * kS);
  viterbi_kernel<<<kB, 128, 0, stream>>>(feats, attn_mask, trans, bp, out);
}
